// RNN_41601053229824
// MI455X (gfx1250) — matrix operand order repaired and checked by reading, not compiled, not run on hardware
//
#include <hip/hip_runtime.h>
#include <math.h>

typedef __attribute__((ext_vector_type(16))) _Float16 v16h;
typedef __attribute__((ext_vector_type(8)))  _Float16 v8h;
typedef __attribute__((ext_vector_type(8)))  float    v8f;

constexpr int BATCH = 65536;
constexpr int SEQ   = 14;
constexpr int INSZ  = 24;
constexpr int NCLS  = 10;
constexpr int HSTR  = 72;           // h-buffer row stride in halves (bank stagger, 16B aligned)
constexpr int ROWS_PER_WAVE = 16;
constexpr int ROWS_PER_BLK  = 128;  // 8 waves * 16 rows

// ---- branchless hardware tanh ----
__device__ __forceinline__ float fast_tanh(float x) {
#if __has_builtin(__builtin_amdgcn_tanhf)
    return __builtin_amdgcn_tanhf(x);          // single v_tanh_f32
#else
    // tanh(x) = sign(x) * (1 - e) / (1 + e),  e = exp(-2|x|) in (0,1] -> no overflow, no branches
    float a = __builtin_fabsf(x);
    float e = __builtin_amdgcn_exp2f(a * -2.885390081777927f); // -2*log2(e)
    float r = __builtin_amdgcn_rcpf(1.0f + e);
    return __builtin_copysignf((1.0f - e) * r, x);
#endif
}

__device__ __forceinline__ v8f wmma_f16(v16h a, v16h b, v8f c) {
    // D = A(16x32 f16) * B(32x16 f16) + C(16x16 f32)
    return __builtin_amdgcn_wmma_f32_16x16x32_f16(false, a, false, b, (short)0, c, false, false);
}

// A-fragment (16x32 f16) from LDS row-major [16][HSTR] buffer.
// ISA layout: lanes 0-15 -> M=lane, K {kt*32+0..7, kt*32+16..23}
//             lanes16-31 -> M=lane-16, K {kt*32+8..15, kt*32+24..31}
__device__ __forceinline__ v16h ldsA(const _Float16* buf, int lane, int kt) {
    const _Float16* row = buf + (lane & 15) * HSTR;
    int g0 = kt * 32 + ((lane & 16) ? 8  : 0);
    int g1 = kt * 32 + ((lane & 16) ? 24 : 16);
    union { v16h v; v8h h[2]; } u;
    u.h[0] = *(const v8h*)(row + g0);
    u.h[1] = *(const v8h*)(row + g1);
    return u.v;
}

// B-fragment (32x16 f16) for D = A*W^T : B[k][n] = W[n][k], W row-major in LDS.
// lanes 0-15 -> N=lane, K {kt*32+0..7, kt*32+16..23} ; lanes 16-31 -> N=lane-16, K {kt*32+8..15, kt*32+24..31}
__device__ __forceinline__ v16h ldsB(const _Float16* w, int lane, int n, int stride, int kt) {
    int k = kt * 32 + ((lane & 16) ? 8 : 0);
    const _Float16* p = w + n * stride + k;
    union { v16h v; v8h h[2]; } u;
    u.h[0] = *(const v8h*)(p);
    u.h[1] = *(const v8h*)(p + 16);
    return u.v;
}

// x_t A-fragment straight from global (K=24 zero-padded to 32)
__device__ __forceinline__ v16h loadX(const float* __restrict__ x, int row, int t, int lane) {
    const float* xp = x + ((size_t)row * SEQ + (size_t)t) * INSZ;
    union { v16h v; _Float16 e[16]; } u;
    if (lane < 16) {
        float4 a = *(const float4*)(xp);
        float4 b = *(const float4*)(xp + 4);
        float4 c = *(const float4*)(xp + 16);
        float4 d = *(const float4*)(xp + 20);
        u.e[0]=(_Float16)a.x; u.e[1]=(_Float16)a.y; u.e[2]=(_Float16)a.z; u.e[3]=(_Float16)a.w;
        u.e[4]=(_Float16)b.x; u.e[5]=(_Float16)b.y; u.e[6]=(_Float16)b.z; u.e[7]=(_Float16)b.w;
        u.e[8]=(_Float16)c.x; u.e[9]=(_Float16)c.y; u.e[10]=(_Float16)c.z; u.e[11]=(_Float16)c.w;
        u.e[12]=(_Float16)d.x; u.e[13]=(_Float16)d.y; u.e[14]=(_Float16)d.z; u.e[15]=(_Float16)d.w;
    } else {
        float4 a = *(const float4*)(xp + 8);
        float4 b = *(const float4*)(xp + 12);
        u.e[0]=(_Float16)a.x; u.e[1]=(_Float16)a.y; u.e[2]=(_Float16)a.z; u.e[3]=(_Float16)a.w;
        u.e[4]=(_Float16)b.x; u.e[5]=(_Float16)b.y; u.e[6]=(_Float16)b.z; u.e[7]=(_Float16)b.w;
        #pragma unroll
        for (int i = 8; i < 16; ++i) u.e[i] = (_Float16)0.f;
    }
    return u.v;
}

// tanh + scatter D tile back into LDS h buffer (D layout: VGPR v -> M = v + 8*(lane>=16), N = nt*16 + lane%16)
__device__ __forceinline__ void storeTanh(_Float16* hbuf, int lane, int nt, v8f acc) {
    int mB = (lane >> 4) << 3;
    _Float16* dst = hbuf + nt * 16 + (lane & 15);
    #pragma unroll
    for (int v = 0; v < 8; ++v)
        dst[(mB + v) * HSTR] = (_Float16)fast_tanh(acc[v]);
}

// Generic hidden layer step (input K = 64): hcur = tanh(hin @ Wih^T + hcur @ Whh^T + bias)
__device__ __forceinline__ void layer64(int lane, const _Float16* hin, _Float16* hcur,
                                        const _Float16* Wih, const _Float16* Whh,
                                        const float* bias) {
    v16h ax0 = ldsA(hin,  lane, 0), ax1 = ldsA(hin,  lane, 1);
    v16h ah0 = ldsA(hcur, lane, 0), ah1 = ldsA(hcur, lane, 1);
    __builtin_amdgcn_wave_barrier();   // A frags fully read before hcur is overwritten
    #pragma unroll
    for (int nt = 0; nt < 4; ++nt) {
        int   n = nt * 16 + (lane & 15);
        float bn = bias[n];
        v8f acc = {bn, bn, bn, bn, bn, bn, bn, bn};
        acc = wmma_f16(ax0, ldsB(Wih, lane, n, 64, 0), acc);
        acc = wmma_f16(ax1, ldsB(Wih, lane, n, 64, 1), acc);
        acc = wmma_f16(ah0, ldsB(Whh, lane, n, 64, 0), acc);
        acc = wmma_f16(ah1, ldsB(Whh, lane, n, 64, 1), acc);
        storeTanh(hcur, lane, nt, acc);
    }
    __builtin_amdgcn_wave_barrier();
}

__global__ __launch_bounds__(256) void rnn_fused(
    const float* __restrict__ x,    const float* __restrict__ h0g,
    const float* __restrict__ Wih0, const float* __restrict__ Whh0,
    const float* __restrict__ bih0, const float* __restrict__ bhh0,
    const float* __restrict__ Wih1, const float* __restrict__ Whh1,
    const float* __restrict__ bih1, const float* __restrict__ bhh1,
    const float* __restrict__ Wih2, const float* __restrict__ Whh2,
    const float* __restrict__ bih2, const float* __restrict__ bhh2,
    const float* __restrict__ Woutg, const float* __restrict__ boutg,
    float* __restrict__ out)
{
    __shared__ __align__(64) _Float16 sWih0[64 * 32];        // K padded 24->32
    __shared__ __align__(64) _Float16 sWhh0[64 * 64];
    __shared__ __align__(64) _Float16 sWih1[64 * 64];
    __shared__ __align__(64) _Float16 sWhh1[64 * 64];
    __shared__ __align__(64) _Float16 sWih2[64 * 64];
    __shared__ __align__(64) _Float16 sWhh2[64 * 64];
    __shared__ float sBias[3][64];                            // b_ih + b_hh fused
    __shared__ float sWout[NCLS * 64];
    __shared__ float sBout[NCLS];
    __shared__ __align__(64) _Float16 sH[8][3][ROWS_PER_WAVE * HSTR];  // per-wave recurrent state

    const int tid  = threadIdx.x;
    const int w    = tid >> 5;
    const int lane = tid & 31;
    const int rowBase = blockIdx.x * ROWS_PER_BLK + w * ROWS_PER_WAVE;

    // ---- stage weights (f32 -> f16) into LDS ----
    for (int i = tid; i < 64 * 32; i += 256) {
        int n = i >> 5, k = i & 31;
        sWih0[i] = (_Float16)((k < INSZ) ? Wih0[n * INSZ + k] : 0.f);
    }
    for (int i = tid; i < 64 * 64; i += 256) {
        sWhh0[i] = (_Float16)Whh0[i];
        sWih1[i] = (_Float16)Wih1[i];
        sWhh1[i] = (_Float16)Whh1[i];
        sWih2[i] = (_Float16)Wih2[i];
        sWhh2[i] = (_Float16)Whh2[i];
    }
    if (tid < 64) {
        sBias[0][tid] = bih0[tid] + bhh0[tid];
        sBias[1][tid] = bih1[tid] + bhh1[tid];
        sBias[2][tid] = bih2[tid] + bhh2[tid];
    }
    for (int i = tid; i < NCLS * 64; i += 256) sWout[i] = Woutg[i];
    if (tid < NCLS) sBout[tid] = boutg[tid];

    // ---- init recurrent state from h0 ----
    #pragma unroll
    for (int l = 0; l < 3; ++l)
        for (int idx = lane; idx < ROWS_PER_WAVE * 64; idx += 32) {
            int r = idx >> 6, c = idx & 63;
            sH[w][l][r * HSTR + c] =
                (_Float16)h0g[((size_t)l << 22) + ((size_t)(rowBase + r) << 6) + c];
        }
    __syncthreads();

    _Float16* h0b = sH[w][0];
    _Float16* h1b = sH[w][1];
    _Float16* h2b = sH[w][2];

    // ---- fused time loop: all 3 layers, state resident in LDS ----
    for (int t = 0; t < SEQ; ++t) {
        // layer 0: x_t (K=24 pad 32) + h0
        {
            v16h axl = loadX(x, rowBase + (lane & 15), t, lane);
            v16h ah0 = ldsA(h0b, lane, 0), ah1 = ldsA(h0b, lane, 1);
            __builtin_amdgcn_wave_barrier();
            #pragma unroll
            for (int nt = 0; nt < 4; ++nt) {
                int   n = nt * 16 + (lane & 15);
                float bn = sBias[0][n];
                v8f acc = {bn, bn, bn, bn, bn, bn, bn, bn};
                acc = wmma_f16(axl, ldsB(sWih0, lane, n, 32, 0), acc);
                acc = wmma_f16(ah0, ldsB(sWhh0, lane, n, 64, 0), acc);
                acc = wmma_f16(ah1, ldsB(sWhh0, lane, n, 64, 1), acc);
                storeTanh(h0b, lane, nt, acc);
            }
            __builtin_amdgcn_wave_barrier();
        }
        layer64(lane, h0b, h1b, sWih1, sWhh1, sBias[1]);   // layer 1
        layer64(lane, h1b, h2b, sWih2, sWhh2, sBias[2]);   // layer 2
    }

    // ---- head: out[b,c] = h2_last . Wout[c] + bout[c] ----
    for (int idx = lane; idx < ROWS_PER_WAVE * NCLS; idx += 32) {
        int r = idx / NCLS, c = idx % NCLS;
        float sum = sBout[c];
        #pragma unroll
        for (int h = 0; h < 64; ++h)
            sum += (float)h2b[r * HSTR + h] * sWout[c * 64 + h];
        out[(size_t)(rowBase + r) * NCLS + c] = sum;
    }

    // ---- h_state[l,b,h] ----
    float* hs = out + (size_t)BATCH * NCLS;
    #pragma unroll
    for (int l = 0; l < 3; ++l)
        for (int idx = lane; idx < ROWS_PER_WAVE * 64; idx += 32) {
            int r = idx >> 6, c = idx & 63;
            hs[((size_t)l << 22) + ((size_t)(rowBase + r) << 6) + c] =
                (float)sH[w][l][r * HSTR + c];
        }
}

extern "C" void kernel_launch(void* const* d_in, const int* in_sizes, int n_in,
                              void* d_out, int out_size, void* d_ws, size_t ws_size,
                              hipStream_t stream) {
    const float* x     = (const float*)d_in[0];
    const float* h0g   = (const float*)d_in[1];
    const float* Wih0  = (const float*)d_in[2];
    const float* Whh0  = (const float*)d_in[3];
    const float* bih0  = (const float*)d_in[4];
    const float* bhh0  = (const float*)d_in[5];
    const float* Wih1  = (const float*)d_in[6];
    const float* Whh1  = (const float*)d_in[7];
    const float* bih1  = (const float*)d_in[8];
    const float* bhh1  = (const float*)d_in[9];
    const float* Wih2  = (const float*)d_in[10];
    const float* Whh2  = (const float*)d_in[11];
    const float* bih2  = (const float*)d_in[12];
    const float* bhh2  = (const float*)d_in[13];
    const float* Wout  = (const float*)d_in[14];
    const float* bout  = (const float*)d_in[15];
    float* out = (float*)d_out;

    dim3 grid(BATCH / ROWS_PER_BLK);   // 512 blocks
    dim3 block(256);                   // 8 wave32s, 16 batch rows each
    rnn_fused<<<grid, block, 0, stream>>>(x, h0g, Wih0, Whh0, bih0, bhh0,
                                          Wih1, Whh1, bih1, bhh1,
                                          Wih2, Whh2, bih2, bhh2,
                                          Wout, bout, out);
}
